// TemplatePointwiseAttention_44375602102631
// MI455X (gfx1250) — compile-verified, matrix operand order checked by reading, not run on hardware
//
#include <hip/hip_runtime.h>

// ---------------- problem constants ----------------
#define NRR  512
#define NT   4
#define CT   64
#define CZ   128
#define NPIX (NRR * NRR)          // 262144 pixels
#define INF_ 100000.0f
#define QSCALE 0.17677669529663687f   // 1/sqrt(32)

// LDS row pads (8 bf16 = 16B = 4-bank rotation per row) for tiles read
// with per-lane-row fragment addressing.
#define CZP (CZ + 8)              // 136
#define CTP (CT + 8)              // 72

typedef __bf16 bf16;
typedef __attribute__((ext_vector_type(16))) __bf16 v16bf;
typedef __attribute__((ext_vector_type(8)))  __bf16 v8bf;
typedef __attribute__((ext_vector_type(8)))  float  v8f;

// ---- fragment loader ---------------------------------------------------
// A / B 16-bit fragment layout (wave32):
//   lane l (0..15):  row/col = l,     K = {base+0..7, base+16..23}
//   lane l (16..31): row/col = l-16,  K = {base+8..15, base+24..31}
// Caller passes p already offset by (k0 + khi) where khi = (lane>=16)*8.
__device__ inline v16bf load_frag_bf16(const bf16* __restrict__ p) {
    v8bf lo = *(const v8bf*)(p);
    v8bf hi = *(const v8bf*)(p + 16);
    return __builtin_shufflevector(lo, hi, 0,1,2,3,4,5,6,7,8,9,10,11,12,13,14,15);
}

// pack 8 f32 -> 8 bf16 and store 16B
__device__ inline void cvt_store8(bf16* dst, float4 a, float4 b) {
    v8bf o;
    o[0] = (bf16)a.x; o[1] = (bf16)a.y; o[2] = (bf16)a.z; o[3] = (bf16)a.w;
    o[4] = (bf16)b.x; o[5] = (bf16)b.y; o[6] = (bf16)b.z; o[7] = (bf16)b.w;
    *(v8bf*)dst = o;
}

// ---- weight prep: f32 [K][N] -> bf16 transposed [N][K] in workspace ----
// ws layout (bf16 elements): wqT @0 (128*128), wkT @16384 (128*64),
//                            wvT @24576 (128*64), woT @32768 (128*128)
__global__ void prep_weights(const float* __restrict__ wq,
                             const float* __restrict__ wk,
                             const float* __restrict__ wv,
                             const float* __restrict__ wo,
                             bf16* __restrict__ ws) {
    int i = blockIdx.x * blockDim.x + threadIdx.x;
    if (i < 16384) {                               // wqT [128][128]
        int n = i >> 7, k = i & 127;
        ws[i] = (bf16)wq[k * CZ + n];
    } else if (i < 24576) {                        // wkT [128][64]
        int j = i - 16384; int n = j >> 6, k = j & 63;
        ws[i] = (bf16)wk[k * CZ + n];
    } else if (i < 32768) {                        // wvT [128][64]
        int j = i - 24576; int n = j >> 6, k = j & 63;
        ws[i] = (bf16)wv[k * CZ + n];
    } else if (i < 49152) {                        // woT [128][128]
        int j = i - 32768; int n = j >> 7, k = j & 127;
        ws[i] = (bf16)wo[k * CZ + n];
    }
}

// ---- fused template pointwise attention --------------------------------
// One block = 16 pixels; 8 waves, wave w computes feature slice [16w,16w+16).
__global__ __launch_bounds__(256)
void tpa_main(const float* __restrict__ t,      // [NT, NPIX, CT]
              const float* __restrict__ z,      // [NPIX, CZ]
              const float* __restrict__ tmask,  // [NT]
              const bf16*  __restrict__ wqT,    // [CZ][CZ]
              const bf16*  __restrict__ wkT,    // [CZ][CT]
              const bf16*  __restrict__ wvT,    // [CZ][CT]
              const bf16*  __restrict__ woT,    // [CZ][CZ]
              const float* __restrict__ bo,     // [CZ]
              float* __restrict__ out)          // [NPIX, CZ]
{
    __shared__ bf16 s_z[16][CZP];          // staged z slab   (~4.3 KB)
    __shared__ bf16 s_t[NT][16][CTP];      // staged t slab   (~9.0 KB)
    __shared__ bf16 s_q[16][CZ];           // scaled q        ( 4 KB)
    __shared__ bf16 s_k[NT][16][CZ];       // k projection    (16 KB)
    __shared__ bf16 s_v[NT][16][CZ];       // v projection    (16 KB)
    __shared__ bf16 s_o[16][CZP];          // attn output     (~4.3 KB)

    const int tid   = threadIdx.x;
    const int wave  = tid >> 5;
    const int lane  = tid & 31;
    const int lrow  = lane & 15;            // fragment row/col selector
    const int khi   = (lane >> 4) << 3;     // 0 or 8: K half for this lane
    const int mbase = (lane >> 4) << 3;     // 0 or 8: C/D row base
    const int n0    = wave << 4;            // feature slice base
    const size_t pix0 = (size_t)blockIdx.x * 16;

    // ============ Stage 0: cooperative stage + convert (once!) ==========
    {   // z slab: 16 rows x 128 f32 = 256 chunks of 8 -> 1 chunk/thread
        const int row = tid >> 4, c8 = (tid & 15) << 3;
        const float* p = z + (pix0 + (size_t)row) * CZ + c8;
        float4 a = *(const float4*)p;
        float4 b = *(const float4*)(p + 4);
        cvt_store8(&s_z[row][c8], a, b);
        // t slab: 4 tmpl x 16 rows x 64 f32 = 512 chunks -> 2 chunks/thread
#pragma unroll
        for (int it = 0; it < 2; ++it) {
            const int s2  = tid + it * 256;
            const int tt  = s2 >> 7;
            const int rem = s2 & 127;
            const int r2  = rem >> 3;
            const int c   = (rem & 7) << 3;
            const float* q = t + ((size_t)tt * NPIX + pix0 + (size_t)r2) * CT + c;
            float4 a2 = *(const float4*)q;
            float4 b2 = *(const float4*)(q + 4);
            cvt_store8(&s_t[tt][r2][c], a2, b2);
        }
    }
    __syncthreads();

    // ============ Stage A: Q/K/V projections (bf16 WMMA) ================
    {   // Q[16,16] slice = Z[16,128] @ wq[:, n0:n0+16], scaled by 1/sqrt(d)
        v8f acc = {};
#pragma unroll
        for (int ks = 0; ks < 4; ++ks) {
            const int k0 = ks * 32;
            v16bf a = load_frag_bf16(&s_z[lrow][k0 + khi]);
            v16bf b = load_frag_bf16(wqT + (size_t)(n0 + lrow) * CZ + k0 + khi);
            acc = __builtin_amdgcn_wmma_f32_16x16x32_bf16(
                      false, a, false, b, (short)0, acc, false, false);
        }
        const int n = n0 + lrow;
#pragma unroll
        for (int r = 0; r < 8; ++r) s_q[mbase + r][n] = (bf16)(acc[r] * QSCALE);
    }

#pragma unroll
    for (int tt = 0; tt < NT; ++tt) {       // K,V slices per template
        v8f acck = {};
        v8f accv = {};
#pragma unroll
        for (int ks = 0; ks < 2; ++ks) {
            const int k0 = ks * 32;
            v16bf a  = load_frag_bf16(&s_t[tt][lrow][k0 + khi]);
            v16bf bk = load_frag_bf16(wkT + (size_t)(n0 + lrow) * CT + k0 + khi);
            v16bf bv = load_frag_bf16(wvT + (size_t)(n0 + lrow) * CT + k0 + khi);
            acck = __builtin_amdgcn_wmma_f32_16x16x32_bf16(
                       false, a, false, bk, (short)0, acck, false, false);
            accv = __builtin_amdgcn_wmma_f32_16x16x32_bf16(
                       false, a, false, bv, (short)0, accv, false, false);
        }
        const int n = n0 + lrow;
#pragma unroll
        for (int r = 0; r < 8; ++r) {
            s_k[tt][mbase + r][n] = (bf16)acck[r];
            s_v[tt][mbase + r][n] = (bf16)accv[r];
        }
    }
    __syncthreads();

    // ============ Stage B: softmax over templates (VALU) ================
    {   // thread -> (pixel m, 8-column group); head is constant per group
        const int m  = tid >> 4;
        const int j  = tid & 15;
        const int c0 = j << 3;
        const int h  = c0 >> 5;

        float lg[NT];
#pragma unroll
        for (int tt = 0; tt < NT; ++tt) {
            const bf16* qp = &s_q[m][h << 5];
            const bf16* kp = &s_k[tt][m][h << 5];
            float acc = 0.f;
#pragma unroll
            for (int dd = 0; dd < 32; ++dd) acc += (float)qp[dd] * (float)kp[dd];
            lg[tt] = acc + INF_ * (tmask[tt] - 1.f);
        }
        float mx = lg[0];
#pragma unroll
        for (int tt = 1; tt < NT; ++tt) mx = fmaxf(mx, lg[tt]);
        float ex[NT];
        float s = 0.f;
#pragma unroll
        for (int tt = 0; tt < NT; ++tt) { ex[tt] = __expf(lg[tt] - mx); s += ex[tt]; }
        const float inv = 1.f / s;
#pragma unroll
        for (int jj = 0; jj < 8; ++jj) {
            float o = 0.f;
#pragma unroll
            for (int tt = 0; tt < NT; ++tt) o += ex[tt] * (float)s_v[tt][m][c0 + jj];
            s_o[m][c0 + jj] = (bf16)(o * inv);
        }
    }
    __syncthreads();

    // ============ Stage C: output projection + bias =====================
    {
        v8f acc = {};
#pragma unroll
        for (int ks = 0; ks < 4; ++ks) {
            const int k0 = ks * 32;
            v16bf a = load_frag_bf16(&s_o[lrow][k0 + khi]);
            v16bf b = load_frag_bf16(woT + (size_t)(n0 + lrow) * CZ + k0 + khi);
            acc = __builtin_amdgcn_wmma_f32_16x16x32_bf16(
                      false, a, false, b, (short)0, acc, false, false);
        }
        const int n = n0 + lrow;
        const float bias = bo[n];
#pragma unroll
        for (int r = 0; r < 8; ++r)
            out[(pix0 + (size_t)(mbase + r)) * CZ + n] = acc[r] + bias;
    }
}

// ---------------- host-side launch --------------------------------------
extern "C" void kernel_launch(void* const* d_in, const int* in_sizes, int n_in,
                              void* d_out, int out_size, void* d_ws, size_t ws_size,
                              hipStream_t stream) {
    const float* t  = (const float*)d_in[0];
    const float* z  = (const float*)d_in[1];
    const float* tm = (const float*)d_in[2];
    const float* wq = (const float*)d_in[3];
    const float* wk = (const float*)d_in[4];
    const float* wv = (const float*)d_in[5];
    const float* wo = (const float*)d_in[6];
    const float* bo = (const float*)d_in[7];
    float* out = (float*)d_out;
    bf16* ws = (bf16*)d_ws;

    hipLaunchKernelGGL(prep_weights, dim3(192), dim3(256), 0, stream,
                       wq, wk, wv, wo, ws);
    hipLaunchKernelGGL(tpa_main, dim3(NPIX / 16), dim3(256), 0, stream,
                       t, z, tm,
                       ws,                // wqT
                       ws + 16384,        // wkT
                       ws + 24576,        // wvT
                       ws + 32768,        // woT
                       bo, out);
}
